// Dipole_Layer_9216999817543
// MI455X (gfx1250) — matrix operand order repaired and checked by reading, not compiled, not run on hardware
//
#include <hip/hip_runtime.h>
#include <hip/hip_bf16.h>

typedef __attribute__((ext_vector_type(16))) __bf16 v16bf;
typedef __attribute__((ext_vector_type(8)))  float  v8f;
typedef __attribute__((ext_vector_type(4)))  float  v4f;

#define CUTOFF 5.0f
#define PI_F 3.14159265358979323846f

// ---------------------------------------------------------------------------
// Fused 2-layer MLP with swish: q = swish(swish(x@W1+b1)@W2+b2)
// x:[N,128] W1:[128,128] W2:[128,32]  -> q:[N,32]
// Block = 128 threads (4 waves); each wave owns a 16-row M-tile (64 rows/blk).
// Weights staged to LDS in WMMA B-fragment order (bf16); GEMM1 output re-laid
// into A-fragment order via per-wave LDS for GEMM2.
// ---------------------------------------------------------------------------
__global__ void __launch_bounds__(128) mlp_kernel(
    const float* __restrict__ x,
    const float* __restrict__ W1, const float* __restrict__ b1,
    const float* __restrict__ W2, const float* __restrict__ b2,
    float* __restrict__ q, int N)
{
    __shared__ alignas(32) __bf16 sW1f[4 * 8 * 32 * 16];   // 32 KB, frag order
    __shared__ alignas(32) __bf16 sW2f[4 * 2 * 32 * 16];   // 8 KB,  frag order
    __shared__ alignas(32) __bf16 sHf[4][4 * 32 * 16];     // 16 KB, per-wave h
    __shared__ float sB1[128];
    __shared__ float sB2[32];

    const int t     = threadIdx.x;
    const int wave  = t >> 5;
    const int lane  = t & 31;
    const int laneN = lane & 15;
    const int half  = lane >> 4;

    // ---- stage W1 (128x128) -> bf16 B-fragment order ----
    // B layout (32x16 tile): lanes 0-15: elems 0-7 = K 0-7, 8-15 = K 16-23;
    // lanes 16-31: elems 0-7 = K 8-15, 8-15 = K 24-31; col = lane%16.
    for (int i = 0; i < 128; ++i) {          // k = i, n = t (coalesced reads)
        const int k = i, n = t;
        const float w = W1[k * 128 + n];
        const int kb = k >> 5, c = k & 31;
        const int l = (((c >> 3) & 1) << 4) + (n & 15);
        sW1f[((kb * 8 + (n >> 4)) * 32 + l) * 16 + (c & 7) + ((c >> 4) << 3)] = (__bf16)w;
    }
    // ---- stage W2 (128x32) ----
    for (int i = 0; i < 32; ++i) {
        const int e = i * 128 + t;           // coalesced
        const int k = e >> 5, n = e & 31;
        const float w = W2[e];
        const int kb = k >> 5, c = k & 31;
        const int l = (((c >> 3) & 1) << 4) + (n & 15);
        sW2f[((kb * 2 + (n >> 4)) * 32 + l) * 16 + (c & 7) + ((c >> 4) << 3)] = (__bf16)w;
    }
    sB1[t] = b1[t];
    if (t < 32) sB2[t] = b2[t];
    __syncthreads();

    const long m0 = (long)blockIdx.x * 64 + (long)wave * 16;

    // ---- A fragments of x (16x32 bf16 per K-block) ----
    // A layout: lanes 0-15 (row=lane): elems 0-7 = K 0-7, 8-15 = K 16-23;
    //           lanes 16-31 (row=lane-16): elems 0-7 = K 8-15, 8-15 = K 24-31.
    int row = (int)m0 + laneN;
    if (row >= N) row = N - 1;               // clamp: EXEC must stay all-ones
    const float* xrow = x + (long)row * 128;

    v16bf a[4];
    #pragma unroll
    for (int kb = 0; kb < 4; ++kb) {
        const int base0 = kb * 32 + half * 8;
        const v4f f0 = *(const v4f*)(xrow + base0);
        const v4f f1 = *(const v4f*)(xrow + base0 + 4);
        const v4f f2 = *(const v4f*)(xrow + base0 + 16);
        const v4f f3 = *(const v4f*)(xrow + base0 + 20);
        #pragma unroll
        for (int j = 0; j < 4; ++j) {
            a[kb][j]      = (__bf16)f0[j];
            a[kb][4 + j]  = (__bf16)f1[j];
            a[kb][8 + j]  = (__bf16)f2[j];
            a[kb][12 + j] = (__bf16)f3[j];
        }
    }

    // ---- GEMM1: 8 N-tiles x 4 K-steps = 32 WMMAs ----
    v8f zero = {};
    v8f acc[8];
    #pragma unroll
    for (int nb = 0; nb < 8; ++nb) acc[nb] = zero;
    #pragma unroll
    for (int kb = 0; kb < 4; ++kb) {
        #pragma unroll
        for (int nb = 0; nb < 8; ++nb) {
            const v16bf bf = *(const v16bf*)&sW1f[((kb * 8 + nb) * 32 + lane) * 16];
            acc[nb] = __builtin_amdgcn_wmma_f32_16x16x32_bf16(
                false, a[kb], false, bf, (short)0, acc[nb], false, false);
        }
    }

    // ---- bias + swish, write h into per-wave LDS in A-fragment order ----
    // C/D layout: col = lane%16, row = elem + 8*half.
    #pragma unroll
    for (int nb = 0; nb < 8; ++nb) {
        const float bias = sB1[nb * 16 + laneN];
        #pragma unroll
        for (int e = 0; e < 8; ++e) {
            float v = acc[nb][e] + bias;
            v = v / (1.0f + __expf(-v));     // swish
            const int r  = e + half * 8;     // row within 16-row tile
            const int n  = nb * 16 + laneN;  // h column (K of GEMM2)
            const int kb2 = n >> 5, c = n & 31;
            const int l   = (((c >> 3) & 1) << 4) + r;
            const int i2  = (c & 7) + ((c >> 4) << 3);
            sHf[wave][(kb2 * 32 + l) * 16 + i2] = (__bf16)v;
        }
    }
    __builtin_amdgcn_wave_barrier();         // wave-local LDS RAW ordering

    // ---- GEMM2: 2 N-tiles x 4 K-steps = 8 WMMAs ----
    v8f acc2[2];
    acc2[0] = zero; acc2[1] = zero;
    #pragma unroll
    for (int kb = 0; kb < 4; ++kb) {
        const v16bf ha = *(const v16bf*)&sHf[wave][(kb * 32 + lane) * 16];
        #pragma unroll
        for (int nb = 0; nb < 2; ++nb) {
            const v16bf bf = *(const v16bf*)&sW2f[((kb * 2 + nb) * 32 + lane) * 16];
            acc2[nb] = __builtin_amdgcn_wmma_f32_16x16x32_bf16(
                false, ha, false, bf, (short)0, acc2[nb], false, false);
        }
    }

    // ---- bias + swish -> q ----
    #pragma unroll
    for (int nb = 0; nb < 2; ++nb) {
        const float bias = sB2[nb * 16 + laneN];
        #pragma unroll
        for (int e = 0; e < 8; ++e) {
            const long rr = m0 + e + half * 8;
            if (rr < N) {
                float v = acc2[nb][e] + bias;
                v = v / (1.0f + __expf(-v));
                q[rr * 32 + nb * 16 + laneN] = v;
            }
        }
    }
}

// ---------------------------------------------------------------------------
// Edge scatter: one wave per edge, lane = D channel.
// mu[dst,d,:] += c(rij) * q[src,d] * vij ; same for branch 2.
// ---------------------------------------------------------------------------
__global__ void __launch_bounds__(256) edge_kernel(
    const float* __restrict__ rij, const float* __restrict__ vij,
    const int* __restrict__ src, const int* __restrict__ dst,
    const float* __restrict__ q, const float* __restrict__ q2,
    float* __restrict__ mu, float* __restrict__ mu2, int E)
{
    const int wave = threadIdx.x >> 5;
    const int lane = threadIdx.x & 31;
    const long e = (long)blockIdx.x * 8 + wave;
    if (e >= E) return;

    const int s = src[e];
    const int d = dst[e];
    const float r = rij[e];
    float c = 0.0f;
    if (r < CUTOFF) c = 0.5f * (__cosf(r * (PI_F / CUTOFF)) + 1.0f);

    const float vx = vij[e * 3 + 0];
    const float vy = vij[e * 3 + 1];
    const float vz = vij[e * 3 + 2];

    const float qs  = q [(long)s * 32 + lane] * c;   // coalesced 128B gather
    const float q2s = q2[(long)s * 32 + lane] * c;

    float* m  = mu  + ((long)d * 32 + lane) * 3;
    float* m2 = mu2 + ((long)d * 32 + lane) * 3;
    unsafeAtomicAdd(&m[0],  qs * vx);
    unsafeAtomicAdd(&m[1],  qs * vy);
    unsafeAtomicAdd(&m[2],  qs * vz);
    unsafeAtomicAdd(&m2[0], q2s * vx);
    unsafeAtomicAdd(&m2[1], q2s * vy);
    unsafeAtomicAdd(&m2[2], q2s * vz);
}

// ---------------------------------------------------------------------------
// Finalize: mu3 = mu x mu2 ; out = w0*mu + w1*mu2 + w2*mu3 + b_mix
// ---------------------------------------------------------------------------
__global__ void __launch_bounds__(256) finalize_kernel(
    const float* __restrict__ mu, const float* __restrict__ mu2,
    const float* __restrict__ w_mix, const float* __restrict__ b_mix,
    float* __restrict__ out, long total)
{
    const long i = (long)blockIdx.x * 256 + threadIdx.x;
    if (i >= total) return;
    const float ax = mu [i * 3 + 0], ay = mu [i * 3 + 1], az = mu [i * 3 + 2];
    const float bx = mu2[i * 3 + 0], by = mu2[i * 3 + 1], bz = mu2[i * 3 + 2];
    const float cx = ay * bz - az * by;
    const float cy = az * bx - ax * bz;
    const float cz = ax * by - ay * bx;
    const float w0 = w_mix[0], w1 = w_mix[1], w2 = w_mix[2], bb = b_mix[0];
    out[i * 3 + 0] = ax * w0 + bx * w1 + cx * w2 + bb;
    out[i * 3 + 1] = ay * w0 + by * w1 + cy * w2 + bb;
    out[i * 3 + 2] = az * w0 + bz * w1 + cz * w2 + bb;
}

extern "C" void kernel_launch(void* const* d_in, const int* in_sizes, int n_in,
                              void* d_out, int out_size, void* d_ws, size_t ws_size,
                              hipStream_t stream) {
    const float* x     = (const float*)d_in[0];
    const float* rij   = (const float*)d_in[1];
    const float* vij   = (const float*)d_in[2];
    const int*   src   = (const int*)  d_in[3];
    const int*   dst   = (const int*)  d_in[4];
    const float* W1    = (const float*)d_in[5];
    const float* b1    = (const float*)d_in[6];
    const float* W2    = (const float*)d_in[7];
    const float* b2    = (const float*)d_in[8];
    const float* W1b   = (const float*)d_in[9];
    const float* b1b   = (const float*)d_in[10];
    const float* W2b   = (const float*)d_in[11];
    const float* b2b   = (const float*)d_in[12];
    const float* w_mix = (const float*)d_in[13];
    const float* b_mix = (const float*)d_in[14];

    const int N = in_sizes[0] / 128;
    const int E = in_sizes[1];

    float* q   = (float*)d_ws;                 // [N,32]
    float* q2  = q  + (size_t)N * 32;          // [N,32]
    float* mu  = q2 + (size_t)N * 32;          // [N,32,3]
    float* mu2 = mu + (size_t)N * 96;          // [N,32,3]

    hipMemsetAsync(mu, 0, (size_t)N * 96 * 2 * sizeof(float), stream);

    const dim3 gb((N + 63) / 64);
    mlp_kernel<<<gb, 128, 0, stream>>>(x, W1,  b1,  W2,  b2,  q,  N);
    mlp_kernel<<<gb, 128, 0, stream>>>(x, W1b, b1b, W2b, b2b, q2, N);

    edge_kernel<<<(E + 7) / 8, 256, 0, stream>>>(rij, vij, src, dst, q, q2, mu, mu2, E);

    const long total = (long)N * 32;
    finalize_kernel<<<(int)((total + 255) / 256), 256, 0, stream>>>(
        mu, mu2, w_mix, b_mix, (float*)d_out, total);
}
